// CogView3PlusTransformerBlock_81552839017114
// MI455X (gfx1250) — matrix operand order repaired and checked by reading, not compiled, not run on hardware
//
#include <hip/hip_runtime.h>
#include <hip/hip_bf16.h>
#include <stdint.h>

#define S_LEN 1248
#define DIM_  2560
#define HEADS_ 64
#define HD_   40
#define HDP_  64
#define NQKV  7680
#define FF_   10240
#define TEMB_ 512
#define NADA  30720   // 12 * DIM_

typedef __attribute__((ext_vector_type(16))) __bf16          bf16x16;
typedef __attribute__((ext_vector_type(16))) unsigned short  u16x16;
typedef __attribute__((ext_vector_type(8)))  unsigned short  u16x8;
typedef __attribute__((ext_vector_type(8)))  float           f32x8;

__device__ __forceinline__ unsigned short f2bf(float f) {
  union { float f; unsigned int u; } c; c.f = f;
  unsigned int u = c.u;
  unsigned int r = (u + 0x7FFFu + ((u >> 16) & 1u)) >> 16;  // RNE
  return (unsigned short)r;
}

__device__ __forceinline__ u16x16 ld8x2(const unsigned short* a, const unsigned short* b) {
  u16x8 lo = *reinterpret_cast<const u16x8*>(a);
  u16x8 hi = *reinterpret_cast<const u16x8*>(b);
  return __builtin_shufflevector(lo, hi, 0, 1, 2, 3, 4, 5, 6, 7, 8, 9, 10, 11, 12, 13, 14, 15);
}
__device__ __forceinline__ u16x16 ld16(const unsigned short* p) { return ld8x2(p, p + 8); }
__device__ __forceinline__ bf16x16 bc(u16x16 u) { return __builtin_bit_cast(bf16x16, u); }

#if __has_builtin(__builtin_amdgcn_ds_swizzle)
#define XSHUF(v, m) __int_as_float(__builtin_amdgcn_ds_swizzle(__float_as_int(v), (((m) << 10) | 0x1f)))
#else
#define XSHUF(v, m) __shfl_xor((v), (m), 32)
#endif

#define WMMA_BF16(A, B, C) \
  __builtin_amdgcn_wmma_f32_16x16x32_bf16(false, (A), false, (B), (short)0, (C), false, false)

// ---------------- adaLN: ada = silu(emb) @ W + b  (1x512 @ 512x30720) ----------------
__global__ __launch_bounds__(256) void adaln_kernel(const float* __restrict__ emb,
                                                    const float* __restrict__ W,
                                                    const float* __restrict__ b,
                                                    float* __restrict__ ada) {
  __shared__ float se[TEMB_];
  int t = threadIdx.x;
  for (int j = t; j < TEMB_; j += 256) {
    float e = emb[j];
    se[j] = e / (1.0f + __expf(-e));
  }
  __syncthreads();
  int col = blockIdx.x * 256 + t;
  float acc = b[col];
  for (int j = 0; j < TEMB_; ++j) acc = fmaf(se[j], W[(size_t)j * NADA + col], acc);
  ada[col] = acc;
}

// ---------------- LayerNorm (no affine) + adaLN modulate -> bf16 ----------------
__global__ __launch_bounds__(256) void ln_mod_kernel(const float* __restrict__ x,
                                                     const float* __restrict__ ada,
                                                     const int* __restrict__ tlen,
                                                     int shT, int scT, int shI, int scI,
                                                     unsigned short* __restrict__ out) {
  __shared__ float red[256];
  int s = blockIdx.x, t = threadIdx.x;
  const float* row = x + (size_t)s * DIM_;
  float lsum = 0.f, lsq = 0.f;
  for (int c = t; c < DIM_; c += 256) { float v = row[c]; lsum += v; lsq += v * v; }
  red[t] = lsum; __syncthreads();
  for (int o = 128; o > 0; o >>= 1) { if (t < o) red[t] += red[t + o]; __syncthreads(); }
  float mean = red[0] * (1.0f / DIM_);
  __syncthreads();
  red[t] = lsq; __syncthreads();
  for (int o = 128; o > 0; o >>= 1) { if (t < o) red[t] += red[t + o]; __syncthreads(); }
  float var = red[0] * (1.0f / DIM_) - mean * mean;
  float rstd = rsqrtf(var + 1e-6f);
  bool txt = s < tlen[0];
  const float* sh = ada + (size_t)(txt ? shT : shI) * DIM_;
  const float* sc = ada + (size_t)(txt ? scT : scI) * DIM_;
  for (int c = t; c < DIM_; c += 256) {
    float v = (row[c] - mean) * rstd;
    v = v * (1.0f + sc[c]) + sh[c];
    out[(size_t)s * DIM_ + c] = f2bf(v);
  }
}

// ---------------- bf16 WMMA GEMM: C[M,N] = A[M,K](bf16) * B[K,N](f32->bf16) + bias ----
// 256 threads = 8 waves; block tile 32(M) x 256(N), K step 32, double-buffered LDS.
template <int GELU_EPI, int OUT_BF>
__global__ __launch_bounds__(256) void gemm_bf16_kernel(const unsigned short* __restrict__ A,
                                                        const float* __restrict__ B,
                                                        const float* __restrict__ bias,
                                                        float* __restrict__ Cf,
                                                        unsigned short* __restrict__ Cb,
                                                        int M, int N, int K) {
  __shared__ __align__(16) unsigned short lA[2][32][40];
  __shared__ __align__(16) unsigned short lB[2][32][264];
  int t = threadIdx.x;
  int rowBase = blockIdx.y * 32;
  int colBase = blockIdx.x * 256;
  int lane = t & 31, wave = t >> 5;
  int lrow = lane & 15, lhi = lane >> 4, kb = lhi * 8;
  f32x8 a00 = {}, a01 = {}, a10 = {}, a11 = {};
  int ar = t >> 3, ac = (t & 7) * 4;            // A staging: 4 bf16 / thread
  int brow = t >> 5, bcol = (t & 31) * 8;       // B staging: 4 x 8 f32 / thread
  int nsteps = K / 32;

  auto stage = [&](int step, int buf) {
    int k0 = step * 32;
    *reinterpret_cast<uint2*>(&lA[buf][ar][ac]) =
        *reinterpret_cast<const uint2*>(A + (size_t)(rowBase + ar) * K + k0 + ac);
#pragma unroll
    for (int i = 0; i < 4; ++i) {
      int r = brow + i * 8;
      const float* bp = B + (size_t)(k0 + r) * N + colBase + bcol;
      float4 b0 = *reinterpret_cast<const float4*>(bp);
      float4 b1 = *reinterpret_cast<const float4*>(bp + 4);
      uint4 pk;
      pk.x = (unsigned)f2bf(b0.x) | ((unsigned)f2bf(b0.y) << 16);
      pk.y = (unsigned)f2bf(b0.z) | ((unsigned)f2bf(b0.w) << 16);
      pk.z = (unsigned)f2bf(b1.x) | ((unsigned)f2bf(b1.y) << 16);
      pk.w = (unsigned)f2bf(b1.z) | ((unsigned)f2bf(b1.w) << 16);
      *reinterpret_cast<uint4*>(&lB[buf][r][bcol]) = pk;
      if (k0 + 64 < K) __builtin_prefetch(bp + (size_t)64 * N, 0, 1);
    }
  };

  stage(0, 0);
  for (int s = 0; s < nsteps; ++s) {
    int cur = s & 1;
    __syncthreads();
    if (s + 1 < nsteps) stage(s + 1, cur ^ 1);

    const unsigned short* ar0 = &lA[cur][lrow][0];
    const unsigned short* ar1 = &lA[cur][16 + lrow][0];
    bf16x16 af0 = bc(ld8x2(ar0 + kb, ar0 + 16 + kb));
    bf16x16 af1 = bc(ld8x2(ar1 + kb, ar1 + 16 + kb));
    int c0 = wave * 32 + lrow;
    u16x16 bu0, bu1;
#pragma unroll
    for (int e = 0; e < 16; ++e) {
      bu0[e] = lB[cur][(e & 7) + 8 * lhi + 16 * (e >> 3)][c0];
      bu1[e] = lB[cur][(e & 7) + 8 * lhi + 16 * (e >> 3)][c0 + 16];
    }
    bf16x16 bf0 = bc(bu0);
    bf16x16 bf1 = bc(bu1);
    a00 = WMMA_BF16(af0, bf0, a00);
    a10 = WMMA_BF16(af1, bf0, a10);
    a01 = WMMA_BF16(af0, bf1, a01);
    a11 = WMMA_BF16(af1, bf1, a11);
  }

#pragma unroll
  for (int q = 0; q < 4; ++q) {
    f32x8 acc = (q == 0) ? a00 : (q == 1) ? a01 : (q == 2) ? a10 : a11;
    int msub = q >> 1, nsub = q & 1;
#pragma unroll
    for (int i = 0; i < 8; ++i) {
      int r = rowBase + msub * 16 + lhi * 8 + i;
      int c = colBase + wave * 32 + nsub * 16 + lrow;
      float v = acc[i] + bias[c];
      if (GELU_EPI) {
        float x3 = v * v * v;
        v = 0.5f * v * (1.0f + tanhf(0.7978845608028654f * (v + 0.044715f * x3)));
      }
      if (OUT_BF) Cb[(size_t)r * N + c] = f2bf(v);
      else        Cf[(size_t)r * N + c] = v;
    }
  }
}

// ---------------- per-head QK LayerNorm + pack Q/K head-major bf16 (pad 40->64) -------
__global__ __launch_bounds__(256) void qk_ln_pack_kernel(const float* __restrict__ qkv,
                                                         unsigned short* __restrict__ Qp,
                                                         unsigned short* __restrict__ Kp) {
  int idx = blockIdx.x * 256 + threadIdx.x;
  if (idx >= S_LEN * HEADS_) return;
  int s = idx / HEADS_, h = idx % HEADS_;
  const float* q = qkv + (size_t)s * NQKV + h * HD_;
  const float* k = q + DIM_;
  float qs = 0.f, qq = 0.f, ks = 0.f, kk = 0.f;
  for (int d = 0; d < HD_; ++d) {
    float a = q[d]; qs += a; qq += a * a;
    float b = k[d]; ks += b; kk += b * b;
  }
  float qm = qs * (1.0f / HD_), km = ks * (1.0f / HD_);
  float qr = rsqrtf(qq * (1.0f / HD_) - qm * qm + 1e-6f);
  float kr = rsqrtf(kk * (1.0f / HD_) - km * km + 1e-6f);
  const float isq = 0.15811388300841898f;  // 1/sqrt(40), folded into Q
  size_t ob = ((size_t)h * S_LEN + s) * HDP_;
  for (int d = 0; d < HDP_; ++d) {
    unsigned short qv = 0, kv = 0;
    if (d < HD_) {
      qv = f2bf((q[d] - qm) * qr * isq);
      kv = f2bf((k[d] - km) * kr);
    }
    Qp[ob + d] = qv; Kp[ob + d] = kv;
  }
}

// ---------------- V pack: transpose to d-major bf16 Vt[h][64][S]; row 40 = ones -------
__global__ __launch_bounds__(256) void v_pack_kernel(const float* __restrict__ qkv,
                                                     unsigned short* __restrict__ Vt) {
  __shared__ __align__(16) unsigned short lT[32][72];  // [s_local][d]
  int h = blockIdx.y;
  int s0 = blockIdx.x * 32;
  int t = threadIdx.x;
  int si = t >> 3, d8 = (t & 7) * 8;
  uint4 pk = {0, 0, 0, 0};
  if (d8 < HD_) {
    const float* vp = qkv + (size_t)(s0 + si) * NQKV + 2 * DIM_ + h * HD_ + d8;
    float4 a = *reinterpret_cast<const float4*>(vp);
    float4 b = *reinterpret_cast<const float4*>(vp + 4);
    pk.x = (unsigned)f2bf(a.x) | ((unsigned)f2bf(a.y) << 16);
    pk.y = (unsigned)f2bf(a.z) | ((unsigned)f2bf(a.w) << 16);
    pk.z = (unsigned)f2bf(b.x) | ((unsigned)f2bf(b.y) << 16);
    pk.w = (unsigned)f2bf(b.z) | ((unsigned)f2bf(b.w) << 16);
  } else if (d8 == HD_) {
    pk.x = 0x3F80u;  // bf16 1.0 at d==40 -> softmax denominator row
  }
  *reinterpret_cast<uint4*>(&lT[si][d8]) = pk;
  __syncthreads();
  int d = t >> 2, s8 = (t & 3) * 8;
  uint4 w;
  unsigned short e0 = lT[s8 + 0][d], e1 = lT[s8 + 1][d], e2 = lT[s8 + 2][d], e3 = lT[s8 + 3][d];
  unsigned short e4 = lT[s8 + 4][d], e5 = lT[s8 + 5][d], e6 = lT[s8 + 6][d], e7 = lT[s8 + 7][d];
  w.x = (unsigned)e0 | ((unsigned)e1 << 16);
  w.y = (unsigned)e2 | ((unsigned)e3 << 16);
  w.z = (unsigned)e4 | ((unsigned)e5 << 16);
  w.w = (unsigned)e6 | ((unsigned)e7 << 16);
  *reinterpret_cast<uint4*>(Vt + ((size_t)h * HDP_ + d) * S_LEN + s0 + s8) = w;
}

// ---------------- flash attention: 2 waves/block, 32 keys/iter, double-buffered -------
// ctx computed transposed: ctx^T = V^T @ P^T (V^T = A operand, all-vector LDS frags).
__global__ __launch_bounds__(64) void flash_attn_kernel(const unsigned short* __restrict__ Qp,
                                                        const unsigned short* __restrict__ Kp,
                                                        const unsigned short* __restrict__ Vt,
                                                        const float* __restrict__ mask,
                                                        unsigned short* __restrict__ ctx) {
  __shared__ __align__(16) unsigned short lK[2][32][72];
  __shared__ __align__(16) unsigned short lVt[2][64][40];
  __shared__ __align__(16) unsigned short lP[2][16][40];
  __shared__ float lS[2][16];
  int t = threadIdx.x;
  int lane = t & 31, wave = t >> 5;
  int lrow = lane & 15, lhi = lane >> 4, kb = lhi * 8;
  int h = blockIdx.y;
  int qt = blockIdx.x * 2 + wave;
  const unsigned short* Qh  = Qp + (size_t)h * S_LEN * HDP_;
  const unsigned short* Kh  = Kp + (size_t)h * S_LEN * HDP_;
  const unsigned short* VtH = Vt + (size_t)h * HDP_ * S_LEN;

  // Q A-fragments for d in [0,32) and [32,64)
  const unsigned short* qp = Qh + (size_t)(qt * 16 + lrow) * HDP_;
  bf16x16 qf0 = bc(ld16(qp + 16 * lhi));
  bf16x16 qf1 = bc(ld16(qp + 32 + 16 * lhi));

  f32x8 o0 = {}, o1 = {}, o2 = {};
  float mr[8];
#pragma unroll
  for (int i = 0; i < 8; ++i) mr[i] = -3.0e38f;

  auto stage = [&](int step, int buf) {
    int kv = step * 32;
#pragma unroll
    for (int i = 0; i < 4; ++i) {
      int idx = i * 64 + t;
      int r = idx >> 3, c8 = (idx & 7) * 8;
      *reinterpret_cast<uint4*>(&lK[buf][r][c8]) =
          *reinterpret_cast<const uint4*>(Kh + (size_t)(kv + r) * HDP_ + c8);
      int d = idx >> 2, k8 = (idx & 3) * 8;
      *reinterpret_cast<uint4*>(&lVt[buf][d][k8]) =
          *reinterpret_cast<const uint4*>(VtH + (size_t)d * S_LEN + kv + k8);
    }
  };

  const int nsteps = S_LEN / 32;
  stage(0, 0);
  for (int s = 0; s < nsteps; ++s) {
    int cur = s & 1;
    int kv = s * 32;
    __syncthreads();
    if (s + 1 < nsteps) stage(s + 1, cur ^ 1);

    // scores: two 16x16 tiles (keys kv..+15, kv+16..+31)
    f32x8 sa = {}, sb = {};
    {
      bf16x16 ka0 = bc(ld16(&lK[cur][lrow][16 * lhi]));
      bf16x16 ka1 = bc(ld16(&lK[cur][lrow][32 + 16 * lhi]));
      sa = WMMA_BF16(qf0, ka0, sa);
      sa = WMMA_BF16(qf1, ka1, sa);
      bf16x16 kb0 = bc(ld16(&lK[cur][16 + lrow][16 * lhi]));
      bf16x16 kb1 = bc(ld16(&lK[cur][16 + lrow][32 + 16 * lhi]));
      sb = WMMA_BF16(qf0, kb0, sb);
      sb = WMMA_BF16(qf1, kb1, sb);
    }
    const float* mrow = mask + (size_t)(qt * 16 + lhi * 8) * S_LEN + kv + lrow;
#pragma unroll
    for (int i = 0; i < 8; ++i) {
      sa[i] += mrow[(size_t)i * S_LEN];
      sb[i] += mrow[(size_t)i * S_LEN + 16];
    }

    // online softmax: row max via xor butterfly; denominator from V^T ones-row (d=40)
#pragma unroll
    for (int i = 0; i < 8; ++i) {
      float r = fmaxf(sa[i], sb[i]);
      r = fmaxf(r, XSHUF(r, 1));
      r = fmaxf(r, XSHUF(r, 2));
      r = fmaxf(r, XSHUF(r, 4));
      r = fmaxf(r, XSHUF(r, 8));
      float mnew = fmaxf(mr[i], r);
      float scale = __expf(mr[i] - mnew);
      mr[i] = mnew;
      float pa = __expf(sa[i] - mnew);
      float pb = __expf(sb[i] - mnew);
      lP[wave][lhi * 8 + i][lrow]      = f2bf(pa);
      lP[wave][lhi * 8 + i][16 + lrow] = f2bf(pb);
      if (lrow == 0) lS[wave][lhi * 8 + i] = scale;  // per-query rescale, re-indexed below
    }
    // wave-private LDS (lP, lS): per-wave in-order DS ops make store->load safe.
    float scq = lS[wave][lrow];  // rescale for this lane's query column
#pragma unroll
    for (int i = 0; i < 8; ++i) { o0[i] *= scq; o1[i] *= scq; o2[i] *= scq; }

    // P^T as B-fragment (two 8-key runs of lP row, kb and 16+kb); V^T tiles as A-fragments (vector loads)
    bf16x16 pf = bc(ld8x2(&lP[wave][lrow][kb], &lP[wave][lrow][16 + kb]));
#pragma unroll
    for (int n = 0; n < 3; ++n) {
      const unsigned short* vr = &lVt[cur][n * 16 + lrow][0];
      bf16x16 vf = bc(ld8x2(vr + kb, vr + 16 + kb));
      if (n == 0)      o0 = WMMA_BF16(vf, pf, o0);
      else if (n == 1) o1 = WMMA_BF16(vf, pf, o1);
      else             o2 = WMMA_BF16(vf, pf, o2);
    }
  }

  // denominator: ctx^T row d=40 -> o2[0] of upper lane half; share via xor-16 swizzle
  float sw = XSHUF(o2[0], 16);
  float lden = lhi ? o2[0] : sw;
  float inv = 1.0f / lden;
  int qg = qt * 16 + lrow;
  size_t base = (size_t)qg * DIM_ + h * HD_;
#pragma unroll
  for (int n = 0; n < 3; ++n) {
    if (n == 2 && lhi) break;  // d >= 40 not stored
    f32x8 o = (n == 0) ? o0 : (n == 1) ? o1 : o2;
    uint4 pk;
    pk.x = (unsigned)f2bf(o[0] * inv) | ((unsigned)f2bf(o[1] * inv) << 16);
    pk.y = (unsigned)f2bf(o[2] * inv) | ((unsigned)f2bf(o[3] * inv) << 16);
    pk.z = (unsigned)f2bf(o[4] * inv) | ((unsigned)f2bf(o[5] * inv) << 16);
    pk.w = (unsigned)f2bf(o[6] * inv) | ((unsigned)f2bf(o[7] * inv) << 16);
    *reinterpret_cast<uint4*>(ctx + base + n * 16 + lhi * 8) = pk;
  }
}

// ---------------- gated residual: out = x + g[c] * delta ----------------
__global__ __launch_bounds__(256) void residual_gate_kernel(const float* __restrict__ x,
                                                            const float* __restrict__ delta,
                                                            const float* __restrict__ ada,
                                                            const int* __restrict__ tlen,
                                                            int gT, int gI,
                                                            float* __restrict__ out) {
  int idx = blockIdx.x * 256 + threadIdx.x;
  if (idx >= S_LEN * DIM_) return;
  int s = idx / DIM_, c = idx % DIM_;
  const float* g = ada + (size_t)(s < tlen[0] ? gT : gI) * DIM_;
  out[idx] = x[idx] + g[c] * delta[idx];
}

// ---------------- host launch ----------------
extern "C" void kernel_launch(void* const* d_in, const int* in_sizes, int n_in,
                              void* d_out, int out_size, void* d_ws, size_t ws_size,
                              hipStream_t stream) {
  const float* hidden  = (const float*)d_in[0];
  const float* emb     = (const float*)d_in[1];
  const float* mask    = (const float*)d_in[2];
  const float* adaln_w = (const float*)d_in[3];
  const float* adaln_b = (const float*)d_in[4];
  const float* qkv_w   = (const float*)d_in[5];
  const float* qkv_b   = (const float*)d_in[6];
  const float* out_w   = (const float*)d_in[7];
  const float* out_b   = (const float*)d_in[8];
  const float* ff1_w   = (const float*)d_in[9];
  const float* ff1_b   = (const float*)d_in[10];
  const float* ff2_w   = (const float*)d_in[11];
  const float* ff2_b   = (const float*)d_in[12];
  const int*   tlen    = (const int*)d_in[13];
  float* out = (float*)d_out;

  char* ws = (char*)d_ws;
  auto au = [](size_t x) { return (x + 255) & ~(size_t)255; };
  size_t off = 0;
  float* ada = (float*)(ws + off);             off = au(off + (size_t)NADA * 4);
  unsigned short* bfA = (unsigned short*)(ws + off);  // xmod -> ctx -> mi
  off = au(off + (size_t)S_LEN * DIM_ * 2);
  float* qkvBuf = (float*)(ws + off);
  unsigned short* ff1g = (unsigned short*)qkvBuf;     // reuse after QKV consumed
  off = au(off + (size_t)S_LEN * NQKV * 4);
  unsigned short* Qp = (unsigned short*)(ws + off); off = au(off + (size_t)HEADS_ * S_LEN * HDP_ * 2);
  unsigned short* Kp = (unsigned short*)(ws + off); off = au(off + (size_t)HEADS_ * S_LEN * HDP_ * 2);
  unsigned short* Vt = (unsigned short*)(ws + off); off = au(off + (size_t)HEADS_ * S_LEN * HDP_ * 2);
  float* dBuf = (float*)(ws + off);            off = au(off + (size_t)S_LEN * DIM_ * 4);  // attn_out / mo
  float* hBuf = (float*)(ws + off);            off = au(off + (size_t)S_LEN * DIM_ * 4);

  // 1. adaLN vector
  adaln_kernel<<<NADA / 256, 256, 0, stream>>>(emb, adaln_w, adaln_b, ada);
  // 2. LN + modulate (msa): text chunks 6,7; image chunks 0,1
  ln_mod_kernel<<<S_LEN, 256, 0, stream>>>(hidden, ada, tlen, 6, 7, 0, 1, bfA);
  // 3. QKV GEMM
  gemm_bf16_kernel<0, 0><<<dim3(NQKV / 256, S_LEN / 32), 256, 0, stream>>>(
      bfA, qkv_w, qkv_b, qkvBuf, nullptr, S_LEN, NQKV, DIM_);
  // 4. per-head QK LN + pack; V transpose pack (d-major, ones row at d=40)
  qk_ln_pack_kernel<<<(S_LEN * HEADS_ + 255) / 256, 256, 0, stream>>>(qkvBuf, Qp, Kp);
  v_pack_kernel<<<dim3(S_LEN / 32, HEADS_), 256, 0, stream>>>(qkvBuf, Vt);
  // 5. attention -> ctx (bf16, reuses bfA)
  flash_attn_kernel<<<dim3(S_LEN / 32, HEADS_), 64, 0, stream>>>(Qp, Kp, Vt, mask, bfA);
  // 6. out projection
  gemm_bf16_kernel<0, 0><<<dim3(DIM_ / 256, S_LEN / 32), 256, 0, stream>>>(
      bfA, out_w, out_b, dBuf, nullptr, S_LEN, DIM_, DIM_);
  // 7. h = hidden + g_msa * attn_out (text chunk 8, image chunk 2)
  residual_gate_kernel<<<(S_LEN * DIM_ + 255) / 256, 256, 0, stream>>>(
      hidden, dBuf, ada, tlen, 8, 2, hBuf);
  // 8. LN + modulate (mlp): text 9,10; image 3,4
  ln_mod_kernel<<<S_LEN, 256, 0, stream>>>(hBuf, ada, tlen, 9, 10, 3, 4, bfA);
  // 9. FF1 + GELU -> bf16 (reuses qkv buffer)
  gemm_bf16_kernel<1, 1><<<dim3(FF_ / 256, S_LEN / 32), 256, 0, stream>>>(
      bfA, ff1_w, ff1_b, nullptr, ff1g, S_LEN, FF_, DIM_);
  // 10. FF2
  gemm_bf16_kernel<0, 0><<<dim3(DIM_ / 256, S_LEN / 32), 256, 0, stream>>>(
      ff1g, ff2_w, ff2_b, dBuf, nullptr, S_LEN, DIM_, FF_);
  // 11. out = h + g_mlp * mo (text 11, image 5)
  residual_gate_kernel<<<(S_LEN * DIM_ + 255) / 256, 256, 0, stream>>>(
      hBuf, dBuf, ada, tlen, 11, 5, out);
}
